// TFNet3_82514911690933
// MI455X (gfx1250) — matrix operand order repaired and checked by reading, not compiled, not run on hardware
//
#include <hip/hip_runtime.h>
#include <hip/hip_bf16.h>
#include <math.h>

// ---------------------------------------------------------------------------
// TFNet on gfx1250: all GEMMs via v_wmma_f32_16x16x32_f16 (f16 in, f32 acc).
// Pipeline:
//  k_pack     : fold BN into scale/shift, repack weights to f16 GEMM layouts
//  k_conv0    : im2col GEMM K=32 + BN/ReLU + maxpool4          -> x0 f16 [B,254,64]
//  k_conv1    : im2col GEMM K=512 (contiguous rows) + BN/ReLU  -> x1 f16 [B,247,64]
//  k_lstm     : per-dir recurrent GEMM 16x256, K=128, 247 steps-> values/valuesT/hT
//  k_hn_proj  : h_n @ W2^T + b (torch flat-reshape h_n quirk)  -> q f32 [B,128]
//  k_u        : tanh(values@W1^T + W1b + q)                    -> u f16
//  k_score    : u @ V^T + Vb (N padded 200->208)               -> score f32
//  k_softmax  : softmax over T, zero-padded T 247->256         -> awT f16 [B,208,256]
//  k_context  : valuesT(128xT) @ awT^T(Tx208) per batch        -> ctx f16 [208,B,128]
//  k_heads    : grouped per-TF MLP + wave shfl_xor reduction   -> out f32 [B,200]
// Requires ws_size >= ~340 MB.
// ---------------------------------------------------------------------------

typedef _Float16 h16;
typedef __attribute__((ext_vector_type(16))) _Float16 v16h;
typedef __attribute__((ext_vector_type(8)))  _Float16 h8;
typedef __attribute__((ext_vector_type(8)))  float    v8f;

#define BB   512
#define LL   1024
#define EMBD 4
#define CC   64
#define HHU  64
#define NTFD 200
#define NTFP 208
#define DD2  128
#define TTS  247
#define TPAD 256
#define POOL0 254

__device__ __forceinline__ v8f wmma16(v16h a, v16h b, v8f c) {
  return __builtin_amdgcn_wmma_f32_16x16x32_f16(false, a, false, b, (short)0, c,
                                                false, false);
}

// A fragment 16x32 f16 from row-major [16][K], uniform row stride ld (halfs).
// Per lane: M = lane&15, K runs {koff..koff+7, koff+16..koff+23}, koff=(lane>=16)?8:0
__device__ __forceinline__ v16h frag_a(const h16* base, int ld) {
  int lane = threadIdx.x & 31;
  const h16* r = base + (size_t)(lane & 15) * ld + ((lane & 16) ? 8 : 0);
  h8 lo = *(const h8*)r;
  h8 hi = *(const h8*)(r + 16);
  v16h a;
#pragma unroll
  for (int i = 0; i < 8; ++i) { a[i] = lo[i]; a[i + 8] = hi[i]; }
  return a;
}

// A fragment when the caller computed a per-lane row pointer (clamped rows).
__device__ __forceinline__ v16h frag_a_row(const h16* rowp) {
  int lane = threadIdx.x & 31;
  const h16* r = rowp + ((lane & 16) ? 8 : 0);
  h8 lo = *(const h8*)r;
  h8 hi = *(const h8*)(r + 16);
  v16h a;
#pragma unroll
  for (int i = 0; i < 8; ++i) { a[i] = lo[i]; a[i + 8] = hi[i]; }
  return a;
}

// B fragment 32x16 f16 from row-major [N][K] (weights stored [out][in]).
// Per lane: N = lane&15, K runs {koff..koff+7, koff+16..koff+23}, koff=(lane>=16)?8:0
__device__ __forceinline__ v16h frag_b(const h16* base, int ld) {
  int lane = threadIdx.x & 31;
  const h16* r = base + (size_t)(lane & 15) * ld + ((lane & 16) ? 8 : 0);
  h8 lo = *(const h8*)r;
  h8 hi = *(const h8*)(r + 16);
  v16h b;
#pragma unroll
  for (int i = 0; i < 8; ++i) { b[i] = lo[i]; b[i + 8] = hi[i]; }
  return b;
}

// ---------------------------------------------------------------------------
__global__ void k_pack(const float* conv0_w, const float* conv0_b,
                       const float* bn0_g, const float* bn0_b,
                       const float* bn0_m, const float* bn0_v,
                       const float* conv1_w, const float* conv1_b,
                       const float* bn1_g, const float* bn1_b,
                       const float* bn1_m, const float* bn1_v,
                       const float* wih_f, const float* whh_f,
                       const float* bih_f, const float* bhh_f,
                       const float* wih_b, const float* whh_b,
                       const float* bih_b, const float* bhh_b,
                       const float* W1_w, const float* W2_w,
                       const float* V_w, const float* V_b,
                       const float* fc1_w,
                       h16* w0r, float* s0, float* sh0,
                       h16* w1r, float* s1, float* sh1,
                       h16* Wcat, float* bias_lstm,
                       h16* W1h, h16* W2h, h16* Vh, float* Vbp,
                       h16* fc1h) {
  int tid = blockIdx.x * blockDim.x + threadIdx.x;
  int nth = gridDim.x * blockDim.x;
  // conv0 weights: B[n=co][kk=k*4+e] = conv0_w[co][e][k]
  for (int i = tid; i < 64 * 32; i += nth) {
    int co = i >> 5, kk = i & 31, k = kk >> 2, e = kk & 3;
    w0r[i] = (h16)conv0_w[co * 32 + e * 8 + k];
  }
  for (int c = tid; c < 64; c += nth) {
    float s = bn0_g[c] * rsqrtf(bn0_v[c] + 1e-5f);
    s0[c] = s;
    sh0[c] = (conv0_b[c] - bn0_m[c]) * s + bn0_b[c];
    float s2 = bn1_g[c] * rsqrtf(bn1_v[c] + 1e-5f);
    s1[c] = s2;
    sh1[c] = (conv1_b[c] - bn1_m[c]) * s2 + bn1_b[c];
  }
  // conv1 weights: B[n=co][kk=k*64+ci] = conv1_w[co][ci][k]
  for (int i = tid; i < 64 * 512; i += nth) {
    int co = i >> 9, kk = i & 511, k = kk >> 6, ci = kk & 63;
    w1r[i] = (h16)conv1_w[co * 512 + ci * 8 + k];
  }
  // LSTM: Wcat[d][n][0:64]=Wih[n], [n][64:128]=Whh[n]
  for (int i = tid; i < 2 * 256 * 128; i += nth) {
    int d = i >> 15, r = i & 32767;
    int n = r >> 7, j = r & 127;
    const float* wi = d ? wih_b : wih_f;
    const float* wh = d ? whh_b : whh_f;
    Wcat[i] = (h16)((j < 64) ? wi[n * 64 + j] : wh[n * 64 + (j - 64)]);
  }
  for (int i = tid; i < 2 * 256; i += nth) {
    int d = i >> 8, n = i & 255;
    bias_lstm[i] = d ? (bih_b[n] + bhh_b[n]) : (bih_f[n] + bhh_f[n]);
  }
  for (int i = tid; i < 128 * 128; i += nth) {
    W1h[i] = (h16)W1_w[i];
    W2h[i] = (h16)W2_w[i];
  }
  for (int i = tid; i < NTFP * 128; i += nth) {
    int o = i >> 7;
    Vh[i] = (o < NTFD) ? (h16)V_w[i] : (h16)0.f;
  }
  for (int o = tid; o < NTFP; o += nth) Vbp[o] = (o < NTFD) ? V_b[o] : 0.f;
  for (int i = tid; i < NTFD * 64 * 128; i += nth) fc1h[i] = (h16)fc1_w[i];
}

// ---------------------------------------------------------------------------
// conv0 + BN + ReLU + maxpool4. One WMMA (K=32) per 16x16 tile; pool inside D.
// grid (16, B), block 128 (4 waves; wave w -> 16 conv positions, all 64 ch).
__global__ void k_conv0(const float* __restrict__ dna, const h16* __restrict__ w0r,
                        const float* __restrict__ s0, const float* __restrict__ sh0,
                        h16* __restrict__ x0) {
  int b = blockIdx.y;
  int wv = threadIdx.x >> 5;
  int lane = threadIdx.x & 31;
  int cm0 = blockIdx.x * 64 + wv * 16;  // conv-position tile base
  int m = lane & 15;
  int t = cm0 + m;
  if (t > 1016) t = 1016;  // clamp (valid conv positions 0..1016)
  const float* rp = dna + ((size_t)b * LL + t) * EMBD;  // 32 contiguous floats
  int koff = (lane & 16) ? 8 : 0;
  v16h a;
#pragma unroll
  for (int i = 0; i < 8; ++i) {
    a[i]     = (h16)rp[koff + i];
    a[i + 8] = (h16)rp[koff + 16 + i];
  }
  int prow = (cm0 >> 2) + ((lane & 16) ? 2 : 0);
#pragma unroll
  for (int nt = 0; nt < 4; ++nt) {
    v8f acc = {};
    v16h bf = frag_b(w0r + nt * 16 * 32, 32);
    acc = wmma16(a, bf, acc);
    int ch = nt * 16 + (lane & 15);
    float s = s0[ch], sh = sh0[ch];
    float y0 = 0.f, y1 = 0.f;  // ReLU makes 0 a safe identity for max
#pragma unroll
    for (int r = 0; r < 4; ++r) {
      y0 = fmaxf(y0, fmaxf(acc[r] * s + sh, 0.f));
      y1 = fmaxf(y1, fmaxf(acc[4 + r] * s + sh, 0.f));
    }
    if (prow < POOL0)     x0[((size_t)b * POOL0 + prow) * CC + ch] = (h16)y0;
    if (prow + 1 < POOL0) x0[((size_t)b * POOL0 + prow + 1) * CC + ch] = (h16)y1;
  }
}

// ---------------------------------------------------------------------------
// conv1 as im2col GEMM: A row (t) = x0[b][t..t+7][0..63] = 512 contiguous halfs.
// grid (16, B), block 128 (wave w -> channel tile w), 16 K-tiles.
__global__ void k_conv1(const h16* __restrict__ x0, const h16* __restrict__ w1r,
                        const float* __restrict__ s1, const float* __restrict__ sh1,
                        h16* __restrict__ x1) {
  int b = blockIdx.y;
  int wv = threadIdx.x >> 5;
  int lane = threadIdx.x & 31;
  int t0 = blockIdx.x * 16;
  int m = lane & 15;
  int te = t0 + m;
  if (te > TTS - 1) te = TTS - 1;
  const h16* rowp = x0 + ((size_t)b * POOL0 + te) * CC;
  int n0 = wv * 16;
  v8f acc = {};
#pragma unroll
  for (int kt = 0; kt < 16; ++kt) {
    v16h a = frag_a_row(rowp + kt * 32);
    v16h bf = frag_b(w1r + (size_t)n0 * 512 + kt * 32, 512);
    acc = wmma16(a, bf, acc);
  }
  int ch = n0 + (lane & 15);
  float s = s1[ch], sh = sh1[ch];
  int roff = (lane & 16) ? 8 : 0;
#pragma unroll
  for (int r = 0; r < 8; ++r) {
    int t = t0 + r + roff;
    if (t < TTS)
      x1[((size_t)b * TTS + t) * CC + ch] = (h16)fmaxf(acc[r] * s + sh, 0.f);
  }
}

// ---------------------------------------------------------------------------
// Bi-LSTM: grid (B/16, 2 dirs), block 256 (8 waves, 2 gate tiles each).
// Per step: gates(16x256) = [x_t|h](16x128) @ Wcat^T, pointwise in LDS.
__global__ void k_lstm(const h16* __restrict__ x1, const h16* __restrict__ Wcat,
                       const float* __restrict__ bias_lstm,
                       h16* __restrict__ values, h16* __restrict__ valuesT,
                       h16* __restrict__ hT) {
  __shared__ __align__(16) h16 At[16][128];   // [row][0:64]=x_t, [64:128]=h
  __shared__ float gbuf[16][260];             // padded stride
  __shared__ float cbuf[16][64];
  __shared__ float biasl[256];
  int dir = blockIdx.y;
  int b0 = blockIdx.x * 16;
  int tid = threadIdx.x;
  int wv = tid >> 5, lane = tid & 31;
  const h16* W = Wcat + (size_t)dir * 256 * 128;

  for (int i = tid; i < 256; i += 256) biasl[i] = bias_lstm[dir * 256 + i];
  for (int i = tid; i < 16 * 64; i += 256) {
    int row = i >> 6, uu = i & 63;
    cbuf[row][uu] = 0.f;
    At[row][64 + uu] = (h16)0.f;  // h0 = 0
    // zero valuesT time padding (t 247..255) so the context GEMM K-pad is clean
    size_t vtrow = ((size_t)(b0 + row) * DD2 + dir * HHU + uu) * TPAD;
    for (int tt = TTS; tt < TPAD; ++tt) valuesT[vtrow + tt] = (h16)0.f;
  }
  __syncthreads();

  for (int s = 0; s < TTS; ++s) {
    int tsrc = dir ? (TTS - 1 - s) : s;
    if (tid < 128) {  // 16 rows x 64 halfs = 128 x uint4
      int row = tid >> 3, off = (tid & 7) * 8;
      *(uint4*)&At[row][off] =
          *(const uint4*)&x1[(((size_t)(b0 + row)) * TTS + tsrc) * CC + off];
    }
    __syncthreads();
#pragma unroll
    for (int g = 0; g < 2; ++g) {
      int nt = wv * 2 + g;
      int n = nt * 16 + (lane & 15);
      float bv = biasl[n];
      v8f acc = {bv, bv, bv, bv, bv, bv, bv, bv};
#pragma unroll
      for (int kt = 0; kt < 4; ++kt) {
        v16h a = frag_a(&At[0][0] + kt * 32, 128);
        v16h bf = frag_b(W + (size_t)(nt * 16) * 128 + kt * 32, 128);
        acc = wmma16(a, bf, acc);
      }
      int roff = (lane & 16) ? 8 : 0;
#pragma unroll
      for (int r = 0; r < 8; ++r) gbuf[r + roff][n] = acc[r];
    }
    __syncthreads();
#pragma unroll
    for (int j = 0; j < 4; ++j) {
      int e = tid + 256 * j;
      int row = e >> 6, uu = e & 63;
      float gi = gbuf[row][uu];
      float gf = gbuf[row][64 + uu];
      float gg = gbuf[row][128 + uu];
      float go = gbuf[row][192 + uu];
      float c = cbuf[row][uu];
      float si = 1.f / (1.f + __expf(-gi));
      float sf = 1.f / (1.f + __expf(-gf));
      float so = 1.f / (1.f + __expf(-go));
      c = sf * c + si * tanhf(gg);
      float h = so * tanhf(c);
      cbuf[row][uu] = c;
      At[row][64 + uu] = (h16)h;
      size_t bb = (size_t)(b0 + row);
      values[(bb * TTS + tsrc) * DD2 + dir * HHU + uu] = (h16)h;
      valuesT[(bb * DD2 + dir * HHU + uu) * TPAD + tsrc] = (h16)h;
      if (s == TTS - 1) hT[((size_t)dir * BB + bb) * HHU + uu] = (h16)h;
    }
    __syncthreads();
  }
}

// ---------------------------------------------------------------------------
// q = h_n @ W2^T + W2_b, replicating torch h_n.view quirk:
// h_n[b][k] = hTflat[(2b + (k>=64))*64 + (k&63)], hTflat = [hTf;hTb] rows.
// grid 32, block 128 (4 waves x 2 N-tiles).
__global__ void k_hn_proj(const h16* __restrict__ hT, const h16* __restrict__ W2h,
                          const float* __restrict__ W2_b, float* __restrict__ q) {
  int wv = threadIdx.x >> 5, lane = threadIdx.x & 31;
  int m0 = blockIdx.x * 16;
  int bq = m0 + (lane & 15);
  int koff = (lane & 16) ? 8 : 0;
#pragma unroll
  for (int g = 0; g < 2; ++g) {
    int nt = wv * 2 + g;
    int n = nt * 16 + (lane & 15);
    float bv = W2_b[n];
    v8f acc = {bv, bv, bv, bv, bv, bv, bv, bv};
#pragma unroll
    for (int kt = 0; kt < 4; ++kt) {
      int k1 = kt * 32 + koff;
      int k2 = k1 + 16;
      const h16* p1 = hT + (size_t)(2 * bq + (k1 >= 64 ? 1 : 0)) * 64 + (k1 & 63);
      const h16* p2 = hT + (size_t)(2 * bq + (k2 >= 64 ? 1 : 0)) * 64 + (k2 & 63);
      h8 lo = *(const h8*)p1;
      h8 hi = *(const h8*)p2;
      v16h a;
#pragma unroll
      for (int i = 0; i < 8; ++i) { a[i] = lo[i]; a[i + 8] = hi[i]; }
      v16h bf = frag_b(W2h + (size_t)(nt * 16) * 128 + kt * 32, 128);
      acc = wmma16(a, bf, acc);
    }
    int roff = (lane & 16) ? 8 : 0;
#pragma unroll
    for (int r = 0; r < 8; ++r)
      q[(size_t)(m0 + r + roff) * DD2 + n] = acc[r];
  }
}

// ---------------------------------------------------------------------------
// u = tanh(values @ W1^T + W1_b + q[b]); grid (16, B), block 128.
__global__ void k_u(const h16* __restrict__ values, const h16* __restrict__ W1h,
                    const float* __restrict__ W1_b, const float* __restrict__ qv,
                    h16* __restrict__ u) {
  int b = blockIdx.y;
  int wv = threadIdx.x >> 5, lane = threadIdx.x & 31;
  int t0 = blockIdx.x * 16;
  int m = lane & 15;
  int te = t0 + m;
  if (te > TTS - 1) te = TTS - 1;
  const h16* rowp = values + ((size_t)b * TTS + te) * DD2;
#pragma unroll
  for (int g = 0; g < 2; ++g) {
    int nt = wv * 2 + g;
    int n = nt * 16 + (lane & 15);
    float bv = W1_b[n] + qv[(size_t)b * DD2 + n];
    v8f acc = {bv, bv, bv, bv, bv, bv, bv, bv};
#pragma unroll
    for (int kt = 0; kt < 4; ++kt) {
      v16h a = frag_a_row(rowp + kt * 32);
      v16h bf = frag_b(W1h + (size_t)(nt * 16) * 128 + kt * 32, 128);
      acc = wmma16(a, bf, acc);
    }
    int roff = (lane & 16) ? 8 : 0;
#pragma unroll
    for (int r = 0; r < 8; ++r) {
      int t = t0 + r + roff;
      if (t < TTS) u[((size_t)b * TTS + t) * DD2 + n] = (h16)tanhf(acc[r]);
    }
  }
}

// ---------------------------------------------------------------------------
// score = u @ V^T + V_b (N padded to 208); grid (16, B), block 128.
__global__ void k_score(const h16* __restrict__ u, const h16* __restrict__ Vh,
                        const float* __restrict__ Vbp, float* __restrict__ score) {
  int b = blockIdx.y;
  int wv = threadIdx.x >> 5, lane = threadIdx.x & 31;
  int t0 = blockIdx.x * 16;
  int m = lane & 15;
  int te = t0 + m;
  if (te > TTS - 1) te = TTS - 1;
  const h16* rowp = u + ((size_t)b * TTS + te) * DD2;
  for (int nt = wv; nt < 13; nt += 4) {
    int n = nt * 16 + (lane & 15);
    float bv = Vbp[n];
    v8f acc = {bv, bv, bv, bv, bv, bv, bv, bv};
#pragma unroll
    for (int kt = 0; kt < 4; ++kt) {
      v16h a = frag_a_row(rowp + kt * 32);
      v16h bf = frag_b(Vh + (size_t)(nt * 16) * 128 + kt * 32, 128);
      acc = wmma16(a, bf, acc);
    }
    int roff = (lane & 16) ? 8 : 0;
#pragma unroll
    for (int r = 0; r < 8; ++r) {
      int t = t0 + r + roff;
      if (t < TTS) score[((size_t)b * TTS + t) * NTFP + n] = acc[r];
    }
  }
}

// ---------------------------------------------------------------------------
// softmax over time; writes awT f16 [B][208][256] with time pad zeroed.
// grid (B, 200), block 256.
__global__ void k_softmax(const float* __restrict__ score, h16* __restrict__ awT) {
  __shared__ float red[256];
  int b = blockIdx.x, k = blockIdx.y;
  int tid = threadIdx.x;
  float v = (tid < TTS) ? score[((size_t)b * TTS + tid) * NTFP + k] : -1e30f;
  red[tid] = v;
  __syncthreads();
  for (int s = 128; s > 0; s >>= 1) {
    if (tid < s) red[tid] = fmaxf(red[tid], red[tid + s]);
    __syncthreads();
  }
  float mx = red[0];
  __syncthreads();
  float e = (tid < TTS) ? __expf(v - mx) : 0.f;
  red[tid] = e;
  __syncthreads();
  for (int s = 128; s > 0; s >>= 1) {
    if (tid < s) red[tid] += red[tid + s];
    __syncthreads();
  }
  float inv = 1.f / red[0];
  awT[((size_t)b * NTFP + k) * TPAD + tid] = (h16)(e * inv);
}

// ---------------------------------------------------------------------------
// context[b,k,d] = sum_t values[b,t,d]*aw[b,t,k]; A=valuesT, B=awT, K=256.
// grid (13 k-tiles, B), block 256 (wave -> d-tile). ctx stored [k][b][d].
__global__ void k_context(const h16* __restrict__ valuesT,
                          const h16* __restrict__ awT, h16* __restrict__ ctx) {
  int b = blockIdx.y;
  int k0 = blockIdx.x * 16;
  int wv = threadIdx.x >> 5, lane = threadIdx.x & 31;
  int d0 = wv * 16;
  v8f acc = {};
#pragma unroll
  for (int kt = 0; kt < 8; ++kt) {
    v16h a = frag_a(valuesT + ((size_t)b * DD2 + d0) * TPAD + kt * 32, TPAD);
    v16h bf = frag_b(awT + ((size_t)b * NTFP + k0) * TPAD + kt * 32, TPAD);
    acc = wmma16(a, bf, acc);
  }
  int k = k0 + (lane & 15);
  int roff = (lane & 16) ? 8 : 0;
#pragma unroll
  for (int r = 0; r < 8; ++r) {
    int d = d0 + r + roff;
    ctx[((size_t)k * BB + b) * DD2 + d] = (h16)acc[r];
  }
}

// ---------------------------------------------------------------------------
// grouped heads: h1 = relu(ctx[k] @ fc1_w[k]^T + fc1_b[k]); out = h1 . fc2_w[k].
// grid (B/16, 200), block 128 (wave -> 16 hidden units); shfl_xor row reduce.
__global__ void k_heads(const h16* __restrict__ ctx, const h16* __restrict__ fc1h,
                        const float* __restrict__ fc1_b,
                        const float* __restrict__ fc2_w,
                        const float* __restrict__ fc2_b, float* __restrict__ out) {
  __shared__ float red[4][16];
  int b0 = blockIdx.x * 16;
  int k = blockIdx.y;
  int wv = threadIdx.x >> 5, lane = threadIdx.x & 31;
  int o0 = wv * 16;
  int o = o0 + (lane & 15);
  float bv = fc1_b[(size_t)k * 64 + o];
  v8f acc = {bv, bv, bv, bv, bv, bv, bv, bv};
#pragma unroll
  for (int kt = 0; kt < 4; ++kt) {
    v16h a = frag_a(ctx + ((size_t)k * BB + b0) * DD2 + kt * 32, DD2);
    v16h bf = frag_b(fc1h + ((size_t)k * 64 + o0) * DD2 + kt * 32, DD2);
    acc = wmma16(a, bf, acc);
  }
  float w2 = fc2_w[(size_t)k * 64 + o];
  float pv[8];
#pragma unroll
  for (int r = 0; r < 8; ++r) pv[r] = fmaxf(acc[r], 0.f) * w2;
#pragma unroll
  for (int msk = 1; msk < 16; msk <<= 1) {
#pragma unroll
    for (int r = 0; r < 8; ++r) pv[r] += __shfl_xor(pv[r], msk, 32);
  }
  if ((lane & 15) == 0) {
    int roff = (lane & 16) ? 8 : 0;
#pragma unroll
    for (int r = 0; r < 8; ++r) red[wv][roff + r] = pv[r];
  }
  __syncthreads();
  if (threadIdx.x < 16) {
    int mrow = threadIdx.x;
    float s = red[0][mrow] + red[1][mrow] + red[2][mrow] + red[3][mrow] + fc2_b[k];
    out[(size_t)(b0 + mrow) * NTFD + k] = s;
  }
}

// ---------------------------------------------------------------------------
extern "C" void kernel_launch(void* const* d_in, const int* in_sizes, int n_in,
                              void* d_out, int out_size, void* d_ws, size_t ws_size,
                              hipStream_t stream) {
  (void)in_sizes; (void)n_in; (void)out_size; (void)ws_size;
  const float* DNA_x   = (const float*)d_in[0];
  const float* conv0_w = (const float*)d_in[1];
  const float* conv0_b = (const float*)d_in[2];
  const float* bn0_g   = (const float*)d_in[3];
  const float* bn0_b   = (const float*)d_in[4];
  const float* bn0_m   = (const float*)d_in[5];
  const float* bn0_v   = (const float*)d_in[6];
  const float* conv1_w = (const float*)d_in[7];
  const float* conv1_b = (const float*)d_in[8];
  const float* bn1_g   = (const float*)d_in[9];
  const float* bn1_b   = (const float*)d_in[10];
  const float* bn1_m   = (const float*)d_in[11];
  const float* bn1_v   = (const float*)d_in[12];
  const float* wih_f   = (const float*)d_in[13];
  const float* whh_f   = (const float*)d_in[14];
  const float* bih_f   = (const float*)d_in[15];
  const float* bhh_f   = (const float*)d_in[16];
  const float* wih_b   = (const float*)d_in[17];
  const float* whh_b   = (const float*)d_in[18];
  const float* bih_b   = (const float*)d_in[19];
  const float* bhh_b   = (const float*)d_in[20];
  const float* W1_w    = (const float*)d_in[21];
  const float* W1_b    = (const float*)d_in[22];
  const float* W2_w    = (const float*)d_in[23];
  const float* W2_b    = (const float*)d_in[24];
  const float* V_w     = (const float*)d_in[25];
  const float* V_b     = (const float*)d_in[26];
  const float* fc1_w   = (const float*)d_in[27];
  const float* fc1_b   = (const float*)d_in[28];
  const float* fc2_w   = (const float*)d_in[29];
  const float* fc2_b   = (const float*)d_in[30];
  float* out = (float*)d_out;

  uint8_t* wsp = (uint8_t*)d_ws;
  size_t off = 0;
  auto alloc = [&](size_t bytes) -> void* {
    void* p = wsp + off;
    off = (off + bytes + 255) & ~(size_t)255;
    return p;
  };
  h16*   w0r       = (h16*)  alloc((size_t)64 * 32 * 2);
  float* s0        = (float*)alloc(64 * 4);
  float* sh0       = (float*)alloc(64 * 4);
  h16*   w1r       = (h16*)  alloc((size_t)64 * 512 * 2);
  float* s1        = (float*)alloc(64 * 4);
  float* sh1       = (float*)alloc(64 * 4);
  h16*   Wcat      = (h16*)  alloc((size_t)2 * 256 * 128 * 2);
  float* bias_lstm = (float*)alloc(2 * 256 * 4);
  h16*   W1h       = (h16*)  alloc((size_t)128 * 128 * 2);
  h16*   W2h       = (h16*)  alloc((size_t)128 * 128 * 2);
  h16*   Vh        = (h16*)  alloc((size_t)NTFP * 128 * 2);
  float* Vbp       = (float*)alloc(NTFP * 4);
  h16*   fc1h      = (h16*)  alloc((size_t)NTFD * 64 * 128 * 2);
  h16*   x0        = (h16*)  alloc((size_t)BB * POOL0 * CC * 2);
  h16*   x1        = (h16*)  alloc((size_t)BB * TTS * CC * 2);
  h16*   values    = (h16*)  alloc((size_t)BB * TTS * DD2 * 2);
  h16*   valuesT   = (h16*)  alloc((size_t)BB * DD2 * TPAD * 2);
  h16*   hT        = (h16*)  alloc((size_t)2 * BB * HHU * 2);
  float* q         = (float*)alloc((size_t)BB * DD2 * 4);
  h16*   u         = (h16*)  alloc((size_t)BB * TTS * DD2 * 2);
  float* score     = (float*)alloc((size_t)BB * TTS * NTFP * 4);
  h16*   awT       = (h16*)  alloc((size_t)BB * NTFP * TPAD * 2);
  h16*   ctx       = (h16*)  alloc((size_t)NTFP * BB * DD2 * 2);

  k_pack<<<dim3(256), 256, 0, stream>>>(
      conv0_w, conv0_b, bn0_g, bn0_b, bn0_m, bn0_v,
      conv1_w, conv1_b, bn1_g, bn1_b, bn1_m, bn1_v,
      wih_f, whh_f, bih_f, bhh_f, wih_b, whh_b, bih_b, bhh_b,
      W1_w, W2_w, V_w, V_b, fc1_w,
      w0r, s0, sh0, w1r, s1, sh1, Wcat, bias_lstm, W1h, W2h, Vh, Vbp, fc1h);
  k_conv0<<<dim3(16, BB), 128, 0, stream>>>(DNA_x, w0r, s0, sh0, x0);
  k_conv1<<<dim3(16, BB), 128, 0, stream>>>(x0, w1r, s1, sh1, x1);
  k_lstm<<<dim3(BB / 16, 2), 256, 0, stream>>>(x1, Wcat, bias_lstm,
                                               values, valuesT, hT);
  k_hn_proj<<<dim3(BB / 16), 128, 0, stream>>>(hT, W2h, W2_b, q);
  k_u<<<dim3(16, BB), 128, 0, stream>>>(values, W1h, W1_b, q, u);
  k_score<<<dim3(16, BB), 128, 0, stream>>>(u, Vh, Vbp, score);
  k_softmax<<<dim3(BB, NTFD), 256, 0, stream>>>(score, awT);
  k_context<<<dim3(13, BB), 256, 0, stream>>>(valuesT, awT, ctx);
  k_heads<<<dim3(BB / 16, NTFD), 128, 0, stream>>>(ctx, fc1h, fc1_b,
                                                   fc2_w, fc2_b, out);
}
